// MLA_24008867184766
// MI455X (gfx1250) — matrix operand order repaired and checked by reading, not compiled, not run on hardware
//
#include <hip/hip_runtime.h>
#include <hip/hip_bf16.h>

// ---------------------------------------------------------------------------
// MLA forward for MI455X (gfx1250), wave32 + WMMA bf16 (f32 accumulate).
//
// Roofline: ~230 GFLOP of matmul vs ~250 MB HBM traffic -> need >=20 TFLOPS
// of matrix throughput to stay memory bound. v_wmma_f32_16x16x32_bf16 gives
// 8x the FLOP/instruction of the f32 WMMA (16x16x4), so all GEMMs + attention
// run in bf16 with f32 accumulation; one elementwise pass converts fp32
// operands to bf16 in workspace.
//
// Data movement: GEMM tiles are staged global->LDS with the gfx1250 async
// path (GLOBAL_LOAD_ASYNC_TO_LDS_B128, ASYNCcnt) and double-buffered so the
// next tile's HBM fetch overlaps the current tile's WMMAs.
// ---------------------------------------------------------------------------

typedef __attribute__((ext_vector_type(16))) __bf16 v16bf;
typedef __attribute__((ext_vector_type(8)))  float  v8f;
typedef int v4i __attribute__((vector_size(16)));

typedef __attribute__((address_space(1))) v4i* gv4i_p;  // global int4*
typedef __attribute__((address_space(3))) v4i* lv4i_p;  // LDS int4*

#define DEVINL __device__ __forceinline__

#if defined(__has_builtin)
#if __has_builtin(__builtin_amdgcn_global_load_async_to_lds_b128)
#define USE_ASYNC_LDS 1
#else
#define USE_ASYNC_LDS 0
#endif
#else
#define USE_ASYNC_LDS 0
#endif

DEVINL unsigned short f2bf(float x) {
  unsigned int u = __float_as_uint(x);
  unsigned int r = u + 0x7FFFu + ((u >> 16) & 1u);   // round-to-nearest-even
  return (unsigned short)(r >> 16);
}

DEVINL v8f wmma_bf16(v16bf a, v16bf b, v8f c) {
  // (neg_a, A, neg_b, B, c_mod, C, reuse_a, reuse_b)
  return __builtin_amdgcn_wmma_f32_16x16x32_bf16(false, a, false, b,
                                                 (short)0, c, false, false);
}

// A-operand (16x32, MxK) gather from row-major bf16 tile in LDS.
// ISA layout: lanes 0-15 hold M=lane, K={0..7,16..23}; lanes 16-31 K+=8.
DEVINL v16bf gather_a(const unsigned short* base, int rowStride) {
  int lane = threadIdx.x & 31;
  int m = lane & 15, half = lane >> 4;
  union { v16bf v; unsigned int u[8]; } f;
#pragma unroll
  for (int j = 0; j < 8; ++j) {
    int kb = ((j & 3) << 1) + ((j >> 2) << 4) + (half << 3);
    f.u[j] = *(const unsigned int*)(base + m * rowStride + kb);
  }
  return f.v;
}

// B-operand (32x16, KxN) gather. LDS tile stored as [n][k] row-major so each
// lane reads packed (k,k+1) pairs. Lanes 0-15: N=lane, K={0..7,16..23};
// lanes 16-31: same N, K+=8 (matches gather_a).
DEVINL v16bf gather_b(const unsigned short* base, int rowStride) {
  int lane = threadIdx.x & 31;
  int n = lane & 15, half = lane >> 4;
  union { v16bf v; unsigned int u[8]; } f;
#pragma unroll
  for (int j = 0; j < 8; ++j) {
    int kb = ((j & 3) << 1) + ((j >> 2) << 4) + (half << 3);
    f.u[j] = *(const unsigned int*)(base + n * rowStride + kb);
  }
  return f.v;
}

// ---------------------------------------------------------------------------
// elementwise fp32 -> bf16
// ---------------------------------------------------------------------------
__global__ void f32_to_bf16_kernel(const float* __restrict__ in,
                                   unsigned short* __restrict__ out, long n) {
  long i = (long)blockIdx.x * blockDim.x + threadIdx.x;
  if (i < n) out[i] = f2bf(in[i]);
}

// ---------------------------------------------------------------------------
// Generic WMMA GEMM: C[M,N] = A[M,K] * W[N,K]^T, A/W bf16, C fp32 or bf16.
// One wave computes a 16x64 C tile (A fragment reused over 4 N-tiles).
// Tiles staged via async global->LDS (B128, 16B chunks), double buffered.
// Per k-step: A = 2 async ops/lane, B = 8 -> 10 wave-level ASYNCcnt items.
// ---------------------------------------------------------------------------
#if USE_ASYNC_LDS
DEVINL void stage_tiles_async(const unsigned short* A, const unsigned short* W,
                              unsigned short* As, unsigned short* Bs,
                              int m0, int n0, int K, int k0, int lane) {
  // A tile 16x32 bf16 = 64 x 16B chunks (rows of 4 chunks)
#pragma unroll
  for (int c = 0; c < 2; ++c) {
    int chunk = lane + 32 * c;              // 0..63
    int r = chunk >> 2, col = (chunk & 3) << 3;
    __builtin_amdgcn_global_load_async_to_lds_b128(
        (gv4i_p)(A + (long)(m0 + r) * K + k0 + col),
        (lv4i_p)(As + chunk * 8), 0, 0);
  }
  // B tile 64x32 bf16 = 256 x 16B chunks
#pragma unroll
  for (int c = 0; c < 8; ++c) {
    int chunk = lane + 32 * c;              // 0..255
    int r = chunk >> 2, col = (chunk & 3) << 3;
    __builtin_amdgcn_global_load_async_to_lds_b128(
        (gv4i_p)(W + (long)(n0 + r) * K + k0 + col),
        (lv4i_p)(Bs + chunk * 8), 0, 0);
  }
}
#endif

template <bool OUT_BF16>
__global__ __launch_bounds__(32)
void gemm_wmma(const unsigned short* __restrict__ A,
               const unsigned short* __restrict__ W,
               void* __restrict__ Cv, int M, int N, int K) {
  __shared__ alignas(16) unsigned short As[2][16 * 32];
  __shared__ alignas(16) unsigned short Bs[2][64 * 32];
  int lane = threadIdx.x;
  int n0 = blockIdx.x * 64;
  int m0 = blockIdx.y * 16;

  v8f acc[4];
#pragma unroll
  for (int t = 0; t < 4; ++t)
#pragma unroll
    for (int r = 0; r < 8; ++r) acc[t][r] = 0.0f;

#if USE_ASYNC_LDS
  stage_tiles_async(A, W, As[0], Bs[0], m0, n0, K, 0, lane);
  for (int k0 = 0; k0 < K; k0 += 32) {
    int p = (k0 >> 5) & 1;
    bool more = (k0 + 32) < K;
    // ensure previous iteration's LDS reads (into the buffer we are about to
    // overwrite) have completed before new async writes can land
    asm volatile("s_wait_dscnt 0x0" ::: "memory");
    if (more) {
      stage_tiles_async(A, W, As[1 - p], Bs[1 - p], m0, n0, K, k0 + 32, lane);
      // 10 newer async ops in flight; <=10 outstanding => current tile landed
      asm volatile("s_wait_asynccnt 0xa" ::: "memory");
    } else {
      asm volatile("s_wait_asynccnt 0x0" ::: "memory");
    }
    v16bf af = gather_a(As[p], 32);
#pragma unroll
    for (int t = 0; t < 4; ++t) {
      v16bf bf = gather_b(Bs[p] + t * 16 * 32, 32);
      acc[t] = wmma_bf16(af, bf, acc[t]);
    }
  }
#else
  for (int k0 = 0; k0 < K; k0 += 32) {
    for (int idx = lane; idx < 16 * 32; idx += 32) {
      int r = idx >> 5, c = idx & 31;
      As[0][idx] = A[(long)(m0 + r) * K + k0 + c];
    }
    for (int idx = lane; idx < 64 * 32; idx += 32) {
      int r = idx >> 5, c = idx & 31;
      Bs[0][idx] = W[(long)(n0 + r) * K + k0 + c];
    }
    __syncthreads();
    v16bf af = gather_a(As[0], 32);
#pragma unroll
    for (int t = 0; t < 4; ++t) {
      v16bf bf = gather_b(Bs[0] + t * 16 * 32, 32);
      acc[t] = wmma_bf16(af, bf, acc[t]);
    }
    __syncthreads();
  }
#endif

  // C/D layout: lane -> col n = lane&15; reg r -> row r + 8*(lane>>4)
  int n = lane & 15, half = lane >> 4;
#pragma unroll
  for (int t = 0; t < 4; ++t)
#pragma unroll
    for (int r = 0; r < 8; ++r) {
      long row = m0 + r + 8 * half;
      long col = n0 + t * 16 + n;
      float v = acc[t][r];
      if (OUT_BF16) ((unsigned short*)Cv)[row * N + col] = f2bf(v);
      else          ((float*)Cv)[row * N + col] = v;
    }
}

// ---------------------------------------------------------------------------
// Row LayerNorm (fp32 in, bf16 out). One 256-thread block per row.
// ---------------------------------------------------------------------------
__global__ void layernorm_kernel(const float* __restrict__ in, int istride, int C,
                                 const float* __restrict__ w,
                                 const float* __restrict__ bsh,
                                 unsigned short* __restrict__ out, int ostride) {
  __shared__ float red[256];
  long row = blockIdx.x;
  const float* x = in + row * (long)istride;
  int t = threadIdx.x;

  float s = 0.0f;
  for (int c = t; c < C; c += 256) s += x[c];
  red[t] = s; __syncthreads();
  for (int o = 128; o > 0; o >>= 1) { if (t < o) red[t] += red[t + o]; __syncthreads(); }
  float mu = red[0] / (float)C;
  __syncthreads();

  float v = 0.0f;
  for (int c = t; c < C; c += 256) { float d = x[c] - mu; v += d * d; }
  red[t] = v; __syncthreads();
  for (int o = 128; o > 0; o >>= 1) { if (t < o) red[t] += red[t + o]; __syncthreads(); }
  float rstd = rsqrtf(red[0] / (float)C + 1e-5f);

  unsigned short* y = out + row * (long)ostride;
  for (int c = t; c < C; c += 256)
    y[c] = f2bf((x[c] - mu) * rstd * w[c] + bsh[c]);
}

// ---------------------------------------------------------------------------
// RoPE (dim 64, half-rotation): out = x*cos + rotate_half(x)*sin, bf16 out.
// ---------------------------------------------------------------------------
DEVINL float rope_angle(int pos, int dd) {
  // inv_freq = 10000^(-(2*(dd mod 32))/64); ln(10000)=9.210340371976184
  float inv = __expf(-(float)(2 * (dd & 31)) * (9.210340371976184f / 64.0f));
  return (float)pos * inv;
}

__global__ void rope_q_kernel(const float* __restrict__ in,
                              unsigned short* __restrict__ out,
                              int S, long total) {
  long i = (long)blockIdx.x * blockDim.x + threadIdx.x;
  if (i >= total) return;
  int dd = (int)(i & 63);
  long row = i >> 10;               // H*64 = 1024 cols per row
  int pos = (int)(row % S);
  float ang = rope_angle(pos, dd);
  float x = in[i];
  float y = (dd < 32) ? -in[i + 32] : in[i - 32];
  out[i] = f2bf(x * cosf(ang) + y * sinf(ang));
}

__global__ void rope_k_kernel(const float* __restrict__ kv_raw,  // [M,576]
                              unsigned short* __restrict__ out,  // [M,64]
                              int S, long total) {
  long i = (long)blockIdx.x * blockDim.x + threadIdx.x;
  if (i >= total) return;
  int dd = (int)(i & 63);
  long row = i >> 6;
  int pos = (int)(row % S);
  const float* x = kv_raw + row * 576 + 512;
  float ang = rope_angle(pos, dd);
  float xv = x[dd];
  float y = (dd < 32) ? -x[dd + 32] : x[dd - 32];
  out[i] = f2bf(xv * cosf(ang) + y * sinf(ang));
}

// ---------------------------------------------------------------------------
// Flash attention: one wave per (b, h, 16-query tile). d_qk=192, d_v=128.
// Scores & PV via bf16 WMMA, online softmax in C/D register layout.
// ---------------------------------------------------------------------------
__global__ __launch_bounds__(32)
void mla_attention(const unsigned short* __restrict__ q_nope, // [M, H*128]
                   const unsigned short* __restrict__ q_rope, // [M, H*64] (roped)
                   const unsigned short* __restrict__ kv_dec, // [M, H*256] (k_nope|v)
                   const unsigned short* __restrict__ k_rope, // [M, 64] (roped)
                   unsigned short* __restrict__ attn_out,     // [M, H*128]
                   int B, int S) {
  const int H = 16;
  __shared__ alignas(16) unsigned short Qs[16 * 192];
  __shared__ alignas(16) unsigned short Ks[32 * 192];
  __shared__ alignas(16) unsigned short Vt[128 * 32];  // V^T : [d][key]
  __shared__ alignas(16) unsigned short Ps[16 * 32];

  int lane = threadIdx.x;
  int tile = blockIdx.x;
  int nQT = S / 16;
  int qt = tile % nQT;
  int h  = (tile / nQT) % H;
  int b  = tile / (nQT * H);
  long rowQ = (long)b * S + qt * 16;

  // stage Q tile [16 x 192] = [q_nope | q_rope]
  for (int idx = lane; idx < 16 * 192; idx += 32) {
    int r = idx / 192, c = idx % 192;
    unsigned short v;
    if (c < 128) v = q_nope[(rowQ + r) * (long)(H * 128) + h * 128 + c];
    else         v = q_rope[(rowQ + r) * (long)(H * 64) + h * 64 + (c - 128)];
    Qs[idx] = v;
  }
  __syncthreads();
  v16bf aq[6];
#pragma unroll
  for (int j = 0; j < 6; ++j) aq[j] = gather_a(Qs + j * 32, 192);

  float m_i[8], l_i[8];
  v8f acc[8];
#pragma unroll
  for (int r = 0; r < 8; ++r) { m_i[r] = -1e30f; l_i[r] = 0.0f; }
#pragma unroll
  for (int t = 0; t < 8; ++t)
#pragma unroll
    for (int r = 0; r < 8; ++r) acc[t][r] = 0.0f;

  const float scale = 0.07216878364870322f;  // 1/sqrt(192)

  for (int kt = 0; kt < S / 32; ++kt) {
    long rowK = (long)b * S + kt * 32;
    // stage K tile [32 x 192] = [k_nope | broadcast k_rope]
    for (int idx = lane; idx < 32 * 192; idx += 32) {
      int r = idx / 192, c = idx % 192;
      unsigned short v;
      if (c < 128) v = kv_dec[(rowK + r) * (long)(H * 256) + h * 256 + c];
      else         v = k_rope[(rowK + r) * 64 + (c - 128)];
      Ks[idx] = v;
    }
    // stage V^T [128 x 32]
    for (int idx = lane; idx < 32 * 128; idx += 32) {
      int key = idx >> 7, d = idx & 127;
      Vt[d * 32 + key] = kv_dec[(rowK + key) * (long)(H * 256) + h * 256 + 128 + d];
    }
    __syncthreads();

    // scores: two 16x16 tiles over d=192 (6 k-steps each)
    v8f s0, s1;
#pragma unroll
    for (int r = 0; r < 8; ++r) { s0[r] = 0.0f; s1[r] = 0.0f; }
#pragma unroll
    for (int j = 0; j < 6; ++j) {
      v16bf b0 = gather_b(Ks + j * 32, 192);             // keys 0..15
      v16bf b1 = gather_b(Ks + 16 * 192 + j * 32, 192);  // keys 16..31
      s0 = wmma_bf16(aq[j], b0, s0);
      s1 = wmma_bf16(aq[j], b1, s1);
    }

    // online softmax (row stats per reg; 16-lane reductions stay in half-wave)
    int n = lane & 15, half = lane >> 4;
#pragma unroll
    for (int r = 0; r < 8; ++r) {
      float sa = s0[r] * scale, sb = s1[r] * scale;
      float mx = fmaxf(sa, sb);
      for (int msk = 1; msk < 16; msk <<= 1)
        mx = fmaxf(mx, __shfl_xor(mx, msk, 32));
      float mnew = fmaxf(m_i[r], mx);
      float alpha = __expf(m_i[r] - mnew);
      float p0 = __expf(sa - mnew), p1 = __expf(sb - mnew);
      float rs = p0 + p1;
      for (int msk = 1; msk < 16; msk <<= 1)
        rs += __shfl_xor(rs, msk, 32);
      l_i[r] = l_i[r] * alpha + rs;
      m_i[r] = mnew;
#pragma unroll
      for (int t = 0; t < 8; ++t) acc[t][r] *= alpha;
      // P (bf16) into LDS: D-layout row = r + 8*half, col = lane&15 (+16)
      int row = r + 8 * half;
      Ps[row * 32 + n]      = f2bf(p0);
      Ps[row * 32 + 16 + n] = f2bf(p1);
    }
    __syncthreads();

    // O += P(16x32) * V(32x128)
    v16bf ap = gather_a(Ps, 32);
#pragma unroll
    for (int t = 0; t < 8; ++t) {
      v16bf bv = gather_b(Vt + t * 16 * 32, 32);
      acc[t] = wmma_bf16(ap, bv, acc[t]);
    }
    __syncthreads();
  }

  // epilogue: normalize by l and store bf16 (feeds o-projection GEMM)
  int n = lane & 15, half = lane >> 4;
#pragma unroll
  for (int r = 0; r < 8; ++r) {
    float inv = 1.0f / l_i[r];
    long orow = (rowQ + r + 8 * half) * (long)(H * 128) + h * 128;
#pragma unroll
    for (int t = 0; t < 8; ++t)
      attn_out[orow + t * 16 + n] = f2bf(acc[t][r] * inv);
  }
}

// ---------------------------------------------------------------------------
// launch
// ---------------------------------------------------------------------------
extern "C" void kernel_launch(void* const* d_in, const int* in_sizes, int n_in,
                              void* d_out, int out_size, void* d_ws, size_t ws_size,
                              hipStream_t stream) {
  (void)in_sizes; (void)n_in; (void)out_size; (void)ws_size;
  const int B = 2, S = 2048, M = B * S, H = 16;
  const int DM = 2048, QL = 1536, KVL = 512;

  const float* hidden    = (const float*)d_in[0];
  const float* q_a_w     = (const float*)d_in[1];
  const float* q_a_ln_w  = (const float*)d_in[2];
  const float* q_a_ln_b  = (const float*)d_in[3];
  const float* q_b_w     = (const float*)d_in[4];
  const float* q_rope_w  = (const float*)d_in[5];
  const float* kv_a_w    = (const float*)d_in[6];
  const float* kv_a_ln_w = (const float*)d_in[7];
  const float* kv_a_ln_b = (const float*)d_in[8];
  const float* kv_b_w    = (const float*)d_in[9];
  const float* o_w       = (const float*)d_in[10];
  float* out = (float*)d_out;

  size_t off = 0;
  auto carve = [&](size_t bytes) -> void* {
    void* p = (char*)d_ws + off;
    off += (bytes + 255) & ~(size_t)255;
    return p;
  };
  // bf16 operand copies
  unsigned short* h_bf     = (unsigned short*)carve((size_t)M * DM * 2);
  unsigned short* qaw_bf   = (unsigned short*)carve((size_t)QL * DM * 2);
  unsigned short* qbw_bf   = (unsigned short*)carve((size_t)(H * 128) * QL * 2);
  unsigned short* qrw_bf   = (unsigned short*)carve((size_t)(H * 64) * QL * 2);
  unsigned short* kvaw_bf  = (unsigned short*)carve((size_t)(KVL + 64) * DM * 2);
  unsigned short* kvbw_bf  = (unsigned short*)carve((size_t)(H * 256) * KVL * 2);
  unsigned short* ow_bf    = (unsigned short*)carve((size_t)DM * (H * 128) * 2);
  // activations
  float*          qa_raw    = (float*)carve((size_t)M * QL * 4);
  unsigned short* qlat_bf   = (unsigned short*)carve((size_t)M * QL * 2);
  float*          kv_raw    = (float*)carve((size_t)M * 576 * 4);
  unsigned short* kvlat_bf  = (unsigned short*)carve((size_t)M * KVL * 2);
  unsigned short* krope_bf  = (unsigned short*)carve((size_t)M * 64 * 2);
  unsigned short* qnope_bf  = (unsigned short*)carve((size_t)M * (H * 128) * 2);
  float*          qrope_raw = (float*)carve((size_t)M * (H * 64) * 4);
  unsigned short* qrope_bf  = (unsigned short*)carve((size_t)M * (H * 64) * 2);
  unsigned short* kvdec_bf  = (unsigned short*)carve((size_t)M * (H * 256) * 2);
  unsigned short* attn_bf   = (unsigned short*)carve((size_t)M * (H * 128) * 2);

  auto conv = [&](const float* src, unsigned short* dst, long n) {
    f32_to_bf16_kernel<<<dim3((unsigned)((n + 255) / 256)), dim3(256), 0, stream>>>(src, dst, n);
  };
  conv(hidden,   h_bf,    (long)M * DM);
  conv(q_a_w,    qaw_bf,  (long)QL * DM);
  conv(q_b_w,    qbw_bf,  (long)(H * 128) * QL);
  conv(q_rope_w, qrw_bf,  (long)(H * 64) * QL);
  conv(kv_a_w,   kvaw_bf, (long)(KVL + 64) * DM);
  conv(kv_b_w,   kvbw_bf, (long)(H * 256) * KVL);
  conv(o_w,      ow_bf,   (long)DM * (H * 128));

  dim3 blk32(32), blk256(256);

  // q_a / kv_a projections from hidden
  gemm_wmma<false><<<dim3(QL / 64, M / 16), blk32, 0, stream>>>(h_bf, qaw_bf, qa_raw, M, QL, DM);
  gemm_wmma<false><<<dim3(576 / 64, M / 16), blk32, 0, stream>>>(h_bf, kvaw_bf, kv_raw, M, 576, DM);

  // layernorms (emit bf16)
  layernorm_kernel<<<dim3(M), blk256, 0, stream>>>(qa_raw, QL, QL, q_a_ln_w, q_a_ln_b, qlat_bf, QL);
  layernorm_kernel<<<dim3(M), blk256, 0, stream>>>(kv_raw, 576, KVL, kv_a_ln_w, kv_a_ln_b, kvlat_bf, KVL);

  // k_rope with RoPE applied (per position, head-broadcast)
  {
    long total = (long)M * 64;
    rope_k_kernel<<<dim3((unsigned)((total + 255) / 256)), blk256, 0, stream>>>(kv_raw, krope_bf, S, total);
  }

  // q_b / q_rope / kv_b projections
  gemm_wmma<true><<<dim3((H * 128) / 64, M / 16), blk32, 0, stream>>>(qlat_bf, qbw_bf, qnope_bf, M, H * 128, QL);
  gemm_wmma<false><<<dim3((H * 64) / 64, M / 16), blk32, 0, stream>>>(qlat_bf, qrw_bf, qrope_raw, M, H * 64, QL);
  {
    long total = (long)M * (H * 64);
    rope_q_kernel<<<dim3((unsigned)((total + 255) / 256)), blk256, 0, stream>>>(qrope_raw, qrope_bf, S, total);
  }
  gemm_wmma<true><<<dim3((H * 256) / 64, M / 16), blk32, 0, stream>>>(kvlat_bf, kvbw_bf, kvdec_bf, M, H * 256, KVL);

  // attention
  mla_attention<<<dim3(B * H * (S / 16)), blk32, 0, stream>>>(qnope_bf, qrope_bf, kvdec_bf, krope_bf, attn_bf, B, S);

  // output projection -> fp32 result
  gemm_wmma<false><<<dim3(DM / 64, M / 16), blk32, 0, stream>>>(attn_bf, ow_bf, out, M, DM, H * 128);
}
